// VQVideo_197568496139
// MI455X (gfx1250) — compile-verified, matrix operand order checked by reading, not run on hardware
//
#include <hip/hip_runtime.h>
#include <cstdint>
#include <cstddef>

// ---------------------------------------------------------------------------
// CDNA5 (gfx1250) VQ-video forward pass.
// All GEMMs run on v_wmma_f32_16x16x32_f16, wave32, 8-wave blocks,
// 128x128x32 tiles. Tiles are staged into LDS by the Tensor Data Mover
// (tensor_load_to_lds, TENSORcnt-tracked, double buffered); TDM's pad
// feature inserts 16B per 64B row => LDS row stride 40 halves (bank spread),
// and TDM OOB zero-fill handles M/N tile edges.
// ---------------------------------------------------------------------------

typedef _Float16 half_t;
typedef __attribute__((ext_vector_type(16))) _Float16 v16h;
typedef __attribute__((ext_vector_type(2)))  _Float16 h2;
typedef __attribute__((ext_vector_type(8)))  float    v8f;
typedef __attribute__((ext_vector_type(4)))  unsigned int u32x4;
typedef __attribute__((ext_vector_type(8)))  int i32x8;
typedef __attribute__((ext_vector_type(4)))  int i32x4;

#if defined(__has_builtin)
#if __has_builtin(__builtin_amdgcn_tensor_load_to_lds)
#define HAVE_TDM 1
#endif
#endif
#ifndef HAVE_TDM
#define HAVE_TDM 0
#endif

namespace cfg {
constexpr int Bq   = 128;
constexpr int T    = 2;
constexpr int Wd   = 512;
constexpr int Ly   = 8;
constexpr int Hh   = 8;
constexpr int NFT  = 128;
constexpr int NDYN = 16;
constexpr int NEMB = 1024;
constexpr int EDIM = 64;
constexpr int VOCAB= 1024;
constexpr int SED  = 256;
constexpr int SE   = T * (NFT + 1);        // 258 encoder ctx
constexpr int SD   = NFT + 1 + NDYN + 1;   // 146 decoder ctx
constexpr int SEp  = 288;                  // 258 padded to K-step multiple
constexpr int SDp  = 160;                  // 146 padded
constexpr long RE  = (long)Bq * SE;        // 33024 encoder rows
constexpr long RD  = (long)Bq * SD;        // 18688 decoder rows
constexpr long RF  = (long)Bq * T * NFT;   // 32768 frame-token rows
constexpr long NQ  = (long)Bq * NDYN;      // 2048 quantized rows
}

// ============================ utility kernels ==============================

__global__ void cvt_f32_f16_kernel(const float* __restrict__ s, half_t* __restrict__ d, long n) {
  long i = (long)blockIdx.x * 256 + threadIdx.x;
  if (i < n) d[i] = (half_t)s[i];
}
__global__ void fill_f32_kernel(float* p, float v, long n) {
  long i = (long)blockIdx.x * 256 + threadIdx.x;
  if (i < n) p[i] = v;
}
__global__ void fill_f16_kernel(half_t* p, float v, long n) {
  long i = (long)blockIdx.x * 256 + threadIdx.x;
  if (i < n) p[i] = (half_t)v;
}
__global__ void fill_i32_kernel(int* p, int v, long n) {
  long i = (long)blockIdx.x * 256 + threadIdx.x;
  if (i < n) p[i] = v;
}

// ============================ TDM helper ===================================
// 2D tile load: tile_dim0 = 32 f16 (64B), tile_dim1 = 128 rows,
// tensor_dim1 = rows remaining (OOB rows zero-filled),
// LDS padding: +4 DWORDs every 16 DWORDs -> row stride 80B = 40 halves.

#if HAVE_TDM
__device__ __forceinline__ void tdm_load_tile(const half_t* gptr, unsigned lds_byte_addr,
                                              int rows_remaining, int lda_elems)
{
  unsigned long long ga = (unsigned long long)(const void*)gptr;
  u32x4 g0;
  g0[0] = 1u;                                              // count=1, user descriptor
  g0[1] = lds_byte_addr;                                   // D#.lds_addr
  g0[2] = (unsigned)(ga & 0xFFFFFFFFu);                    // global_addr[31:0]
  g0[3] = (unsigned)((ga >> 32) & 0x01FFFFFFu) | (2u << 30); // addr[56:32], type=2
  i32x8 g1;
  g1[0] = (1 << 16)    // data_size = 2 bytes
        | (1 << 20)    // pad_enable
        | (3 << 22)    // pad_interval: 16 DWORDs (64B)
        | (3 << 25);   // pad_amount:   4 DWORDs (16B)
  g1[1] = (32 << 16);                                      // tensor_dim0 = 32 (lo16)
  g1[2] = ((unsigned)rows_remaining & 0xFFFFu) << 16;      // tensor_dim1 lo16
  g1[3] = (((unsigned)rows_remaining >> 16) & 0xFFFFu)     // tensor_dim1 hi16
        | (32 << 16);                                      // tile_dim0 = 32
  g1[4] = 128;                                             // tile_dim1 = 128, tile_dim2 = 0
  g1[5] = lda_elems;                                       // tensor_dim0_stride lo32
  g1[6] = 0;
  g1[7] = 0;
  i32x4 z4 = {0, 0, 0, 0};
#if __clang_major__ >= 23
  i32x8 z8 = {0, 0, 0, 0, 0, 0, 0, 0};
  __builtin_amdgcn_tensor_load_to_lds(g0, g1, z4, z4, z8, 0);
#else
  __builtin_amdgcn_tensor_load_to_lds(g0, g1, z4, z4, 0);
#endif
}
#endif

// ============================ WMMA GEMM ====================================
// C[m,n] = epi( sum_k A[m,k] * Bw[n,k] )   (A: MxK f16, Bw: NxK f16 row-major)
// flags: 1=+bias  2=+residual  4=quickGELU  8=store f32  16=store f16
// Batched via blockIdx.z; z -> (zb = z/hdiv, zh = z%hdiv), per-operand strides.

union FragH { v16h v; h2 p[8]; };

__device__ __forceinline__ void mma_step(const half_t* AsP, const half_t* BsP,
                                         int wm, int wn, int mrow, int khalf,
                                         v8f acc[2][4])
{
  // A fragment: 16-bit A 16x32 layout (ISA 7.12.2): lane m = lane%16,
  // VGPR v<4 -> K = 2v + 8*(lane/16); v>=4 -> K = 16 + 2(v-4) + 8*(lane/16).
  FragH af[2], bf[4];
  #pragma unroll
  for (int mt = 0; mt < 2; ++mt) {
    const int r = wm * 32 + mt * 16 + mrow;
    #pragma unroll
    for (int v = 0; v < 8; ++v) {
      const int k = ((v < 4) ? (v * 2) : (16 + (v - 4) * 2)) + khalf;
      af[mt].p[v] = *(const h2*)(&AsP[r * 40 + k]);
    }
  }
  #pragma unroll
  for (int nt = 0; nt < 4; ++nt) {
    const int r = wn * 64 + nt * 16 + mrow;
    #pragma unroll
    for (int v = 0; v < 8; ++v) {
      const int k = ((v < 4) ? (v * 2) : (16 + (v - 4) * 2)) + khalf;
      bf[nt].p[v] = *(const h2*)(&BsP[r * 40 + k]);
    }
  }
  #pragma unroll
  for (int mt = 0; mt < 2; ++mt)
    #pragma unroll
    for (int nt = 0; nt < 4; ++nt)
      acc[mt][nt] = __builtin_amdgcn_wmma_f32_16x16x32_f16(
          false, af[mt].v, false, bf[nt].v, (short)0, acc[mt][nt], false, false);
}

__global__ __launch_bounds__(256)
void gemm_wmma_kernel(const half_t* __restrict__ A, long sAb, long sAh, int lda,
                      const half_t* __restrict__ Bw, long sBb, long sBh, int ldb,
                      float* __restrict__ C32, half_t* __restrict__ C16,
                      long sCb, long sCh, int ldc,
                      const float* __restrict__ bias,
                      const float* __restrict__ Res, long sRb, long sRh, int ldr,
                      int M, int N, int K, int hdiv, int flags)
{
  __shared__ half_t As[2][128 * 40];   // 128 rows x 32 halves + TDM pad (40-half stride)
  __shared__ half_t Bs[2][128 * 40];

  const int tid  = threadIdx.x;
  const int lane = tid & 31;
  const int wave = tid >> 5;        // 8 waves
  const int wm   = wave & 3;        // 4 wave-rows of 32
  const int wn   = wave >> 2;       // 2 wave-cols of 64
  const int zb   = blockIdx.z / hdiv;
  const int zh   = blockIdx.z % hdiv;
  const int tileM = blockIdx.y * 128;
  const int tileN = blockIdx.x * 128;

  v8f acc[2][4];
  for (int i = 0; i < 2; ++i)
    for (int j = 0; j < 4; ++j)
      for (int r = 0; r < 8; ++r) acc[i][j][r] = 0.0f;

  const int mrow  = lane & 15;
  const int khalf = (lane >> 4) << 3;

  const half_t* Abase = A + (zb * sAb + zh * sAh) + (long)tileM * lda;
  const half_t* Bbase = Bw + (zb * sBb + zh * sBh) + (long)tileN * ldb;
  const int remM = M - tileM;       // >=1 (grid covers M); TDM zero-fills beyond
  const int remN = N - tileN;

#if HAVE_TDM
  // --- TDM double-buffered pipeline ---
  if (tid < 32) {                   // one issue per step; tensor ops ignore EXEC
    tdm_load_tile(Abase, (unsigned)(uintptr_t)&As[0][0], remM, lda);
    tdm_load_tile(Bbase, (unsigned)(uintptr_t)&Bs[0][0], remN, ldb);
  }
  int p = 0;
  for (int kb = 0; kb < K; kb += 32) {
    if (tid < 32) {
      if (kb + 32 < K) {            // prefetch next pair into the other buffer
        tdm_load_tile(Abase + kb + 32, (unsigned)(uintptr_t)&As[p ^ 1][0], remM, lda);
        tdm_load_tile(Bbase + kb + 32, (unsigned)(uintptr_t)&Bs[p ^ 1][0], remN, ldb);
        __builtin_amdgcn_s_wait_tensorcnt(2);   // oldest pair (current buffer) done
      } else {
        __builtin_amdgcn_s_wait_tensorcnt(0);
      }
    }
    __syncthreads();                // publish LDS tile to all 8 waves
    mma_step(&As[p][0], &Bs[p][0], wm, wn, mrow, khalf, acc);
    __syncthreads();                // protect buffer p from next TDM overwrite
    p ^= 1;
  }
#else
  // --- fallback: per-thread global->VGPR->LDS staging (round-1 path) ---
  const int ldRow = tid >> 1;
  const int ldSeg = (tid & 1) << 1;
  const int gmA = tileM + ldRow;
  const int gnB = tileN + ldRow;
  const half_t* pA = A + (zb * sAb + zh * sAh) + (long)(gmA < M ? gmA : 0) * lda;
  const half_t* pB = Bw + (zb * sBb + zh * sBh) + (long)(gnB < N ? gnB : 0) * ldb;
  for (int kb = 0; kb < K; kb += 32) {
    uint4 a0 = {0u,0u,0u,0u}, a1 = {0u,0u,0u,0u};
    uint4 b0 = {0u,0u,0u,0u}, b1 = {0u,0u,0u,0u};
    if (gmA < M) {
      a0 = *(const uint4*)(pA + kb + ldSeg * 8);
      a1 = *(const uint4*)(pA + kb + ldSeg * 8 + 8);
    }
    if (gnB < N) {
      b0 = *(const uint4*)(pB + kb + ldSeg * 8);
      b1 = *(const uint4*)(pB + kb + ldSeg * 8 + 8);
    }
    *(uint4*)(&As[0][ldRow * 40 + ldSeg * 8])     = a0;
    *(uint4*)(&As[0][ldRow * 40 + ldSeg * 8 + 8]) = a1;
    *(uint4*)(&Bs[0][ldRow * 40 + ldSeg * 8])     = b0;
    *(uint4*)(&Bs[0][ldRow * 40 + ldSeg * 8 + 8]) = b1;
    __syncthreads();
    mma_step(&As[0][0], &Bs[0][0], wm, wn, mrow, khalf, acc);
    __syncthreads();
  }
#endif

  // Epilogue. C/D layout (ISA 7.12.2): VGPR r -> M = r + 8*(lane/16), N = lane%16.
  const long cOff = zb * sCb + zh * sCh;
  const long rOff = zb * sRb + zh * sRh;
  const int mmAdd = (lane >> 4) << 3;
  const int nl    = lane & 15;
  #pragma unroll
  for (int mt = 0; mt < 2; ++mt)
    #pragma unroll
    for (int nt = 0; nt < 4; ++nt)
      #pragma unroll
      for (int r = 0; r < 8; ++r) {
        const int m = tileM + wm * 32 + mt * 16 + r + mmAdd;
        const int n = tileN + wn * 64 + nt * 16 + nl;
        if (m < M && n < N) {
          float v = acc[mt][nt][r];
          if (flags & 1)  v += bias[n];
          if (flags & 4)  v = v / (1.0f + __expf(-1.702f * v));   // quickGELU
          if (flags & 2)  v += Res[rOff + (long)m * ldr + n];
          if (flags & 8)  C32[cOff + (long)m * ldc + n] = v;
          if (flags & 16) C16[cOff + (long)m * ldc + n] = (half_t)v;
        }
      }
}

// ============================ model kernels ================================

__global__ __launch_bounds__(256)
void ln_kernel(const float* __restrict__ X, const float* __restrict__ g,
               const float* __restrict__ b, half_t* __restrict__ O)
{
  __shared__ float red[256];
  const long row = blockIdx.x;
  const int tid = threadIdx.x;
  const float* x = X + row * 512;
  const float v0 = x[tid], v1 = x[tid + 256];
  red[tid] = v0 + v1; __syncthreads();
  for (int s = 128; s; s >>= 1) { if (tid < s) red[tid] += red[tid + s]; __syncthreads(); }
  const float mean = red[0] * (1.0f / 512.0f);
  __syncthreads();
  const float d0 = v0 - mean, d1 = v1 - mean;
  red[tid] = d0 * d0 + d1 * d1; __syncthreads();
  for (int s = 128; s; s >>= 1) { if (tid < s) red[tid] += red[tid + s]; __syncthreads(); }
  const float rstd = rsqrtf(red[0] * (1.0f / 512.0f) + 1e-5f);
  O[row * 512 + tid]       = (half_t)(d0 * rstd * g[tid] + b[tid]);
  O[row * 512 + 256 + tid] = (half_t)(d1 * rstd * g[tid + 256] + b[tid + 256]);
}

__global__ void split_qkv_kernel(const half_t* __restrict__ qkv,
                                 half_t* __restrict__ q, half_t* __restrict__ k,
                                 half_t* __restrict__ vT, int S, int Sp)
{
  long i = (long)blockIdx.x * 256 + threadIdx.x;        // over B*S*512
  long total = (long)cfg::Bq * S * 512;
  if (i >= total) return;
  int c = (int)(i & 511);
  long rs = i >> 9;
  int s = (int)(rs % S);
  int b = (int)(rs / S);
  int h = c >> 6, d = c & 63;
  long src = rs * 1536;
  long zd  = ((long)(b * cfg::Hh + h) * S + s) * 64 + d;
  q[zd] = qkv[src + c];
  k[zd] = qkv[src + 512 + c];
  vT[((long)(b * cfg::Hh + h) * 64 + d) * Sp + s] = qkv[src + 1024 + c];
}

__global__ __launch_bounds__(256)
void softmax_kernel(const float* __restrict__ S32, half_t* __restrict__ O,
                    int Slen, int Sp, float scale)
{
  __shared__ float red[256];
  const long row = blockIdx.x;
  const int tid = threadIdx.x;
  const float* src = S32 + row * Slen;
  half_t* dst = O + row * Sp;
  float mx = -1e30f;
  for (int i = tid; i < Slen; i += 256) mx = fmaxf(mx, src[i] * scale);
  red[tid] = mx; __syncthreads();
  for (int s = 128; s; s >>= 1) { if (tid < s) red[tid] = fmaxf(red[tid], red[tid + s]); __syncthreads(); }
  mx = red[0]; __syncthreads();
  float sum = 0.0f;
  for (int i = tid; i < Slen; i += 256) sum += __expf(src[i] * scale - mx);
  red[tid] = sum; __syncthreads();
  for (int s = 128; s; s >>= 1) { if (tid < s) red[tid] += red[tid + s]; __syncthreads(); }
  const float inv = 1.0f / red[0];
  for (int i = tid; i < Sp; i += 256)
    dst[i] = (half_t)((i < Slen) ? __expf(src[i] * scale - mx) * inv : 0.0f);
}

__global__ void gather_embed_kernel(const int* __restrict__ x,
                                    const float* __restrict__ table,
                                    half_t* __restrict__ out)
{
  const long row = blockIdx.x;                       // B*T*NFT rows
  const int c = threadIdx.x;                         // SED=256
  out[row * 256 + c] = (half_t)table[(long)x[row] * 256 + c];
}

__global__ void assemble_enc_kernel(const float* __restrict__ embs,
                                    const float* __restrict__ delim,
                                    const float* __restrict__ pos,
                                    float* __restrict__ X)
{
  const long r = blockIdx.x;                         // B*258
  const int b = (int)(r / cfg::SE), s = (int)(r % cfg::SE);
  const int t = s / (cfg::NFT + 1), p = s % (cfg::NFT + 1);
  for (int c = threadIdx.x; c < 512; c += 256) {
    float v = (p == cfg::NFT) ? delim[c]
                              : embs[((long)(b * cfg::T + t) * cfg::NFT + p) * 512 + c];
    X[r * 512 + c] = v + pos[(long)s * 512 + c];
  }
}

__global__ void assemble_dec_kernel(const float* __restrict__ embs,
                                    const float* __restrict__ delim,
                                    const float* __restrict__ fW,
                                    const float* __restrict__ pos,
                                    float* __restrict__ X)
{
  const long r = blockIdx.x;                         // B*146
  const int b = (int)(r / cfg::SD), s = (int)(r % cfg::SD);
  for (int c = threadIdx.x; c < 512; c += 256) {
    float v;
    if (s < cfg::NFT)                 v = embs[((long)(b * cfg::T) * cfg::NFT + s) * 512 + c];
    else if (s == cfg::NFT || s == cfg::SD - 1) v = delim[c];
    else                              v = fW[((long)b * cfg::NDYN + (s - cfg::NFT - 1)) * 512 + c];
    X[r * 512 + c] = v + pos[(long)s * 512 + c];
  }
}

__global__ void gather_rows_f16_kernel(const float* __restrict__ X, half_t* __restrict__ O,
                                       int srcS, int keep)
{
  const long r = blockIdx.x;
  const int b = (int)(r / keep), s = (int)(r % keep);
  const long src = ((long)b * srcS + s) * 512;
  for (int c = threadIdx.x; c < 512; c += 256)
    O[r * 512 + c] = (half_t)X[src + c];
}

__global__ void cnorm_kernel(const float* __restrict__ cb, float* __restrict__ cn)
{
  int j = blockIdx.x * 256 + threadIdx.x;
  if (j < cfg::NEMB) {
    float s = 0.0f;
    for (int d = 0; d < cfg::EDIM; ++d) { float v = cb[(long)j * 64 + d]; s += v * v; }
    cn[j] = s;
  }
}

__global__ __launch_bounds__(256)
void quant_kernel(const float* __restrict__ dots, const float* __restrict__ cnorm,
                  const float* __restrict__ femb, const float* __restrict__ codebook,
                  float* __restrict__ enc, int* __restrict__ counts,
                  float* __restrict__ lossacc, int* __restrict__ idxout)
{
  __shared__ float bs[256];
  __shared__ int   bi[256];
  const int row = blockIdx.x;
  const int tid = threadIdx.x;
  const float* dr = dots + (long)row * 1024;
  float best = -3.0e38f; int bidx = 0;
  for (int j = tid; j < 1024; j += 256) {
    float s = 2.0f * dr[j] - cnorm[j];       // argmin dist == argmax(2*dot - |c|^2)
    if (s > best) { best = s; bidx = j; }
  }
  bs[tid] = best; bi[tid] = bidx; __syncthreads();
  for (int st = 128; st; st >>= 1) {
    if (tid < st) {
      if (bs[tid + st] > bs[tid] || (bs[tid + st] == bs[tid] && bi[tid + st] < bi[tid])) {
        bs[tid] = bs[tid + st]; bi[tid] = bi[tid + st];
      }
    }
    __syncthreads();
  }
  const int idx = bi[0];
  __syncthreads();
  if (tid == 0) {
    idxout[row] = idx;
    enc[(long)row * 1024 + idx] = 1.0f;
    atomicAdd(counts + idx, 1);
  }
  float part = 0.0f;
  if (tid < 64) {
    float d = codebook[(long)idx * 64 + tid] - femb[(long)row * 64 + tid];
    part = d * d;
  }
  bs[tid] = part; __syncthreads();
  for (int st = 128; st; st >>= 1) { if (tid < st) bs[tid] += bs[tid + st]; __syncthreads(); }
  if (tid == 0) atomicAdd(lossacc, bs[0]);
}

__global__ void gather_code_kernel(const float* __restrict__ cb, const int* __restrict__ idx,
                                   half_t* __restrict__ out)
{
  const int r = blockIdx.x;
  out[(long)r * 64 + threadIdx.x] = (half_t)cb[(long)idx[r] * 64 + threadIdx.x];
}

__global__ __launch_bounds__(256)
void finalize_kernel(const int* __restrict__ counts, const float* __restrict__ lossacc,
                     float* __restrict__ out_latent, float* __restrict__ out_perp)
{
  __shared__ float red[256];
  const int tid = threadIdx.x;
  float s = 0.0f;
  for (int j = tid; j < 1024; j += 256) {
    float p = (float)counts[j] * (1.0f / 2048.0f);
    s += p * logf(p + 1e-10f);
  }
  red[tid] = s; __syncthreads();
  for (int st = 128; st; st >>= 1) { if (tid < st) red[tid] += red[tid + st]; __syncthreads(); }
  if (tid == 0) {
    *out_perp   = expf(-red[0]);
    // forward value of q_loss == e_loss -> latent = (1+CC)*MSE, CC=0.25
    *out_latent = lossacc[0] * 1.25f / (2048.0f * 64.0f);
  }
}

// ============================ host orchestration ===========================

static inline long cdiv(long a, long b) { return (a + b - 1) / b; }

struct TFW {
  const half_t *wq, *wo, *wf, *wp;
  const float *ln1g, *ln1b, *qkvb, *outb, *ln2g, *ln2b, *fcb, *pb;
};

static void launch_gemm(hipStream_t st,
                        const half_t* A, long sAb, long sAh, int lda,
                        const half_t* Bw, long sBb, long sBh, int ldb,
                        float* C32, half_t* C16, long sCb, long sCh, int ldc,
                        const float* bias,
                        const float* Res, long sRb, long sRh, int ldr,
                        int M, int N, int K, int Z, int hdiv, int flags)
{
  dim3 g((unsigned)cdiv(N, 128), (unsigned)cdiv(M, 128), (unsigned)Z);
  gemm_wmma_kernel<<<g, 256, 0, st>>>(A, sAb, sAh, lda, Bw, sBb, sBh, ldb,
                                      C32, C16, sCb, sCh, ldc, bias,
                                      Res, sRb, sRh, ldr, M, N, K, hdiv, flags);
}

static void run_tf(hipStream_t st, float* X, int S, int Sp, const TFW& w,
                   half_t* Hln, half_t* qkv16, half_t* q16, half_t* k16,
                   half_t* vT16, float* scores, half_t* att16, half_t* fc16)
{
  using namespace cfg;
  const long rows = (long)Bq * S;
  const int  BH   = Bq * Hh;

  // zero V^T once so padded K columns contribute nothing
  {
    long n = (long)BH * 64 * Sp;
    fill_f16_kernel<<<(unsigned)cdiv(n, 256), 256, 0, st>>>(vT16, 0.0f, n);
  }

  for (int l = 0; l < Ly; ++l) {
    const half_t* wq = w.wq + (long)l * 3 * Wd * Wd;
    const half_t* wo = w.wo + (long)l * Wd * Wd;
    const half_t* wf = w.wf + (long)l * 4 * Wd * Wd;
    const half_t* wp = w.wp + (long)l * Wd * 4 * Wd;

    ln_kernel<<<(unsigned)rows, 256, 0, st>>>(X, w.ln1g + l * Wd, w.ln1b + l * Wd, Hln);

    // qkv = ln(x) @ Wqkv^T + b   -> f16
    launch_gemm(st, Hln, 0, 0, Wd, wq, 0, 0, Wd,
                nullptr, qkv16, 0, 0, 3 * Wd, w.qkvb + (long)l * 3 * Wd,
                nullptr, 0, 0, 0, (int)rows, 3 * Wd, Wd, 1, 1, 1 | 16);

    split_qkv_kernel<<<(unsigned)cdiv(rows * 512, 256), 256, 0, st>>>(qkv16, q16, k16, vT16, S, Sp);

    // scores[z] = q[z] @ k[z]^T  (batched z = b*H+h)
    launch_gemm(st, q16, (long)S * 64, 0, 64, k16, (long)S * 64, 0, 64,
                scores, nullptr, (long)S * S, 0, S, nullptr,
                nullptr, 0, 0, 0, S, S, 64, BH, 1, 8);

    softmax_kernel<<<(unsigned)((long)BH * S), 256, 0, st>>>(scores, att16, S, Sp, 0.125f);

    // O[b,s,h*64+d] = att[z] @ V[z]  (via V^T), written straight into Hln f16
    launch_gemm(st, att16, (long)Hh * S * Sp, (long)S * Sp, Sp,
                vT16, (long)Hh * 64 * Sp, (long)64 * Sp, Sp,
                nullptr, Hln, (long)S * Wd, 64, Wd, nullptr,
                nullptr, 0, 0, 0, S, 64, Sp, BH, Hh, 16);

    // x += O @ Wout^T + b
    launch_gemm(st, Hln, 0, 0, Wd, wo, 0, 0, Wd,
                X, nullptr, 0, 0, Wd, w.outb + (long)l * Wd,
                X, 0, 0, Wd, (int)rows, Wd, Wd, 1, 1, 1 | 2 | 8);

    ln_kernel<<<(unsigned)rows, 256, 0, st>>>(X, w.ln2g + l * Wd, w.ln2b + l * Wd, Hln);

    // fc = gelu(ln(x) @ Wfc^T + b) -> f16
    launch_gemm(st, Hln, 0, 0, Wd, wf, 0, 0, Wd,
                nullptr, fc16, 0, 0, 4 * Wd, w.fcb + (long)l * 4 * Wd,
                nullptr, 0, 0, 0, (int)rows, 4 * Wd, Wd, 1, 1, 1 | 4 | 16);

    // x += fc @ Wp^T + b
    launch_gemm(st, fc16, 0, 0, 4 * Wd, wp, 0, 0, 4 * Wd,
                X, nullptr, 0, 0, Wd, w.pb + (long)l * Wd,
                X, 0, 0, Wd, (int)rows, Wd, 4 * Wd, 1, 1, 1 | 2 | 8);
  }
}

extern "C" void kernel_launch(void* const* d_in, const int* in_sizes, int n_in,
                              void* d_out, int out_size, void* d_ws, size_t ws_size,
                              hipStream_t stream)
{
  using namespace cfg;
  (void)in_sizes; (void)n_in; (void)out_size; (void)ws_size;

  auto F = [&](int i) { return (const float*)d_in[i]; };
  const int* xin = (const int*)d_in[0];

  // ---- workspace bump allocator ----
  char* base = (char*)d_ws;
  size_t off = 0;
  auto alloc = [&](size_t bytes) -> void* {
    void* p = base + off;
    off = (off + bytes + 255) & ~(size_t)255;
    return p;
  };

  // f16 weights
  half_t* wFrame   = (half_t*)alloc((size_t)Wd * SED * 2);
  half_t* wEncProj = (half_t*)alloc((size_t)EDIM * Wd * 2);
  half_t* wDiff    = (half_t*)alloc((size_t)Wd * EDIM * 2);
  half_t* wCode    = (half_t*)alloc((size_t)NEMB * EDIM * 2);
  half_t* wPred    = (half_t*)alloc((size_t)VOCAB * Wd * 2);
  half_t* wQKV[2], *wOut[2], *wFc[2], *wPw[2];
  for (int t = 0; t < 2; ++t) {
    wQKV[t] = (half_t*)alloc((size_t)Ly * 3 * Wd * Wd * 2);
    wOut[t] = (half_t*)alloc((size_t)Ly * Wd * Wd * 2);
    wFc[t]  = (half_t*)alloc((size_t)Ly * 4 * Wd * Wd * 2);
    wPw[t]  = (half_t*)alloc((size_t)Ly * Wd * 4 * Wd * 2);
  }
  // activations
  half_t* gathSE = (half_t*)alloc((size_t)RF * SED * 2);
  float*  embsF  = (float*) alloc((size_t)RF * Wd * 4);
  float*  Xenc   = (float*) alloc((size_t)RE * Wd * 4);
  float*  Xdec   = (float*) alloc((size_t)RD * Wd * 4);
  half_t* Hln    = (half_t*)alloc((size_t)RE * Wd * 2);
  half_t* qkv16  = (half_t*)alloc((size_t)RE * 3 * Wd * 2);
  half_t* q16    = (half_t*)alloc((size_t)Bq * Hh * SE * 64 * 2);
  half_t* k16    = (half_t*)alloc((size_t)Bq * Hh * SE * 64 * 2);
  half_t* vT16   = (half_t*)alloc((size_t)Bq * Hh * 64 * SEp * 2);
  float*  scores = (float*) alloc((size_t)Bq * Hh * SE * SE * 4);
  half_t* att16  = (half_t*)alloc((size_t)Bq * Hh * SE * SEp * 2);
  half_t* fc16   = (half_t*)alloc((size_t)RE * 4 * Wd * 2);
  half_t* gath16 = (half_t*)alloc((size_t)NQ * Wd * 2);
  half_t* gOut16 = (half_t*)alloc((size_t)Bq * NFT * Wd * 2);
  float*  femb32 = (float*) alloc((size_t)NQ * EDIM * 4);
  half_t* femb16 = (half_t*)alloc((size_t)NQ * EDIM * 2);
  float*  dots   = (float*) alloc((size_t)NQ * NEMB * 4);
  float*  cnorm  = (float*) alloc((size_t)NEMB * 4);
  half_t* fq16   = (half_t*)alloc((size_t)NQ * EDIM * 2);
  float*  fW32   = (float*) alloc((size_t)NQ * Wd * 4);
  int*    idxb   = (int*)   alloc((size_t)NQ * 4);
  int*    counts = (int*)   alloc((size_t)NEMB * 4);
  float*  lossac = (float*) alloc(16);

  // outputs: [logits | latent | perplexity | encodings]
  float* out_logits = (float*)d_out;
  float* out_latent = out_logits + (long)Bq * NFT * VOCAB;
  float* out_perp   = out_latent + 1;
  float* out_enc    = out_perp + 1;

  auto cvt = [&](const float* s, half_t* d, long n) {
    cvt_f32_f16_kernel<<<(unsigned)cdiv(n, 256), 256, 0, stream>>>(s, d, n);
  };

  // ---- weight conversion (f32 -> f16, every call, deterministic) ----
  cvt(F(2),  wFrame,   (long)Wd * SED);
  cvt(F(7),  wEncProj, (long)EDIM * Wd);
  cvt(F(3),  wDiff,    (long)Wd * EDIM);
  cvt(F(4),  wCode,    (long)NEMB * EDIM);
  cvt(F(22), wPred,    (long)VOCAB * Wd);
  cvt(F(10), wQKV[0], (long)Ly * 3 * Wd * Wd);
  cvt(F(12), wOut[0], (long)Ly * Wd * Wd);
  cvt(F(16), wFc[0],  (long)Ly * 4 * Wd * Wd);
  cvt(F(18), wPw[0],  (long)Ly * Wd * 4 * Wd);
  cvt(F(25), wQKV[1], (long)Ly * 3 * Wd * Wd);
  cvt(F(27), wOut[1], (long)Ly * Wd * Wd);
  cvt(F(31), wFc[1],  (long)Ly * 4 * Wd * Wd);
  cvt(F(33), wPw[1],  (long)Ly * Wd * 4 * Wd);
  cnorm_kernel<<<4, 256, 0, stream>>>(F(4), cnorm);

  // ---- encoder ----
  gather_embed_kernel<<<(unsigned)RF, 256, 0, stream>>>(xin, F(1), gathSE);
  launch_gemm(stream, gathSE, 0, 0, SED, wFrame, 0, 0, SED,
              embsF, nullptr, 0, 0, Wd, nullptr, nullptr, 0, 0, 0,
              (int)RF, Wd, SED, 1, 1, 8);
  assemble_enc_kernel<<<(unsigned)RE, 256, 0, stream>>>(embsF, F(5), F(6), Xenc);

  TFW enc = { wQKV[0], wOut[0], wFc[0], wPw[0],
              F(8), F(9), F(11), F(13), F(14), F(15), F(17), F(19) };
  run_tf(stream, Xenc, SE, SEp, enc, Hln, qkv16, q16, k16, vT16, scores, att16, fc16);

  // ---- quantize ----
  gather_rows_f16_kernel<<<(unsigned)NQ, 256, 0, stream>>>(Xenc, gath16, SE, NDYN);
  launch_gemm(stream, gath16, 0, 0, Wd, wEncProj, 0, 0, Wd,
              femb32, femb16, 0, 0, EDIM, nullptr, nullptr, 0, 0, 0,
              (int)NQ, EDIM, Wd, 1, 1, 8 | 16);
  launch_gemm(stream, femb16, 0, 0, EDIM, wCode, 0, 0, EDIM,
              dots, nullptr, 0, 0, NEMB, nullptr, nullptr, 0, 0, 0,
              (int)NQ, NEMB, EDIM, 1, 1, 8);

  fill_f32_kernel<<<(unsigned)cdiv((long)NQ * NEMB, 256), 256, 0, stream>>>(out_enc, 0.0f, (long)NQ * NEMB);
  fill_i32_kernel<<<4, 256, 0, stream>>>(counts, 0, NEMB);
  fill_f32_kernel<<<1, 256, 0, stream>>>(lossac, 0.0f, 1);
  quant_kernel<<<(unsigned)NQ, 256, 0, stream>>>(dots, cnorm, femb32, F(4),
                                                 out_enc, counts, lossac, idxb);
  gather_code_kernel<<<(unsigned)NQ, 64, 0, stream>>>(F(4), idxb, fq16);
  launch_gemm(stream, fq16, 0, 0, EDIM, wDiff, 0, 0, EDIM,
              fW32, nullptr, 0, 0, Wd, nullptr, nullptr, 0, 0, 0,
              (int)NQ, Wd, EDIM, 1, 1, 8);

  // ---- decoder ----
  assemble_dec_kernel<<<(unsigned)RD, 256, 0, stream>>>(embsF, F(20), fW32, F(21), Xdec);
  TFW dec = { wQKV[1], wOut[1], wFc[1], wPw[1],
              F(23), F(24), F(26), F(28), F(29), F(30), F(32), F(34) };
  run_tf(stream, Xdec, SD, SDp, dec, Hln, qkv16, q16, k16, vT16, scores, att16, fc16);

  // ---- prediction head ----
  gather_rows_f16_kernel<<<(unsigned)((long)Bq * NFT), 256, 0, stream>>>(Xdec, gOut16, SD, NFT);
  launch_gemm(stream, gOut16, 0, 0, Wd, wPred, 0, 0, Wd,
              out_logits, nullptr, 0, 0, VOCAB, nullptr, nullptr, 0, 0, 0,
              Bq * NFT, VOCAB, Wd, 1, 1, 8);

  // ---- scalars ----
  finalize_kernel<<<1, 256, 0, stream>>>(counts, lossac, out_latent, out_perp);
}
